// GNN_27650999451833
// MI455X (gfx1250) — matrix operand order repaired and checked by reading, not compiled, not run on hardware
//
#include <hip/hip_runtime.h>
#include <math.h>

// ---------------- types for gfx1250 WMMA (wave32) ----------------
typedef __bf16 bf16_t;
typedef __attribute__((ext_vector_type(16))) __bf16 v16bf;
typedef __attribute__((ext_vector_type(8)))  __bf16 v8bf;
typedef __attribute__((ext_vector_type(8)))  float  v8f;

#define N_NODES 8192
#define N_EDGES 65536

#define WMMA_BF16(A, B, C) \
  __builtin_amdgcn_wmma_f32_16x16x32_bf16(false, (A), false, (B), (short)0, (C), false, false)

__device__ inline v16bf cat8(v8bf x, v8bf y) {
  return __builtin_shufflevector(x, y, 0,1,2,3,4,5,6,7,8,9,10,11,12,13,14,15);
}

// A-fragment (16x32 bf16, ISA layout): lanes 0-15: M=lane, K=0..7 & 16..23
//                                      lanes 16-31: M=lane-16, K=8..15 & 24..31
// seg = pointer to this lane's 32-wide K segment (64B, 16B-aligned)
__device__ inline v16bf load_a_frag(const bf16_t* seg, int hi) {
  const v8bf* p = (const v8bf*)seg;
  return cat8(p[hi], p[2 + hi]);
}

// B-fragment (32x16 bf16): lane&15 = N (row of W^T); lanes 0-15: K=0..7 & 16..23, lanes 16-31: K=8..15 & 24..31
// seg = pointer to this lane's W^T row at current K base
__device__ inline v16bf load_b_frag(const bf16_t* seg, int hi) {
  const v8bf* p = (const v8bf*)seg;
  return cat8(p[hi], p[2 + hi]);
}

// one K-step updating a 16x64 strip (4 accumulators) — branch-free
__device__ inline void gemm_step4(const bf16_t* aseg,
                                  const bf16_t* br0, const bf16_t* br1,
                                  const bf16_t* br2, const bf16_t* br3,
                                  int kb, int hi,
                                  v8f& c0, v8f& c1, v8f& c2, v8f& c3) {
  v16bf a = load_a_frag(aseg, hi);
  c0 = WMMA_BF16(a, load_b_frag(br0 + kb, hi), c0);
  c1 = WMMA_BF16(a, load_b_frag(br1 + kb, hi), c1);
  c2 = WMMA_BF16(a, load_b_frag(br2 + kb, hi), c2);
  c3 = WMMA_BF16(a, load_b_frag(br3 + kb, hi), c3);
}

// order-preserving float<->uint encode for atomic max
__device__ inline unsigned fenc(float f) {
  unsigned u = __float_as_uint(f);
  return (u & 0x80000000u) ? ~u : (u | 0x80000000u);
}
__device__ inline float fdec(unsigned e) {
  unsigned u = (e & 0x80000000u) ? (e ^ 0x80000000u) : ~e;
  return __uint_as_float(u);
}

// ---------------- conversion / init kernels ----------------
__global__ void f2bf_kernel(const float* __restrict__ in, bf16_t* __restrict__ out, int n) {
  int i = blockIdx.x * blockDim.x + threadIdx.x;
  if (i < n) out[i] = (bf16_t)in[i];
}

__global__ void f2bf_pad_kernel(const float* __restrict__ in, bf16_t* __restrict__ out,
                                int rows, int cin, int cout) {
  int i = blockIdx.x * blockDim.x + threadIdx.x;
  if (i >= rows * cout) return;
  int r = i / cout, c = i % cout;
  float v = (c < cin) ? in[(size_t)r * cin + c] : 0.f;
  out[i] = (bf16_t)v;
}

// Wet [1024][2080]: K' 0..1023 <- We rows 0..1023 (src part); 1024..1039 <- s_f rows (+16 zero pad);
// 1056..2079 <- We rows 1040..2063 (dst part). We is [2064,1024] row-major.
__global__ void conv_wet_kernel(const float* __restrict__ We, bf16_t* __restrict__ Wet) {
  int i = blockIdx.x * blockDim.x + threadIdx.x;
  if (i >= 1024 * 2080) return;
  int n = i / 2080, k = i % 2080;
  int r = 0; bool valid = true;
  if (k < 1024) r = k;
  else if (k < 1056) { int kk = k - 1024; if (kk < 16) r = 1024 + kk; else valid = false; }
  else r = 1040 + (k - 1056);
  Wet[i] = (bf16_t)(valid ? We[(size_t)r * 1024 + n] : 0.f);
}

// Plain transpose: Wnt [1024][2048] <- Wn [2048][1024]
__global__ void conv_wnt_kernel(const float* __restrict__ Wn, bf16_t* __restrict__ Wnt) {
  int i = blockIdx.x * blockDim.x + threadIdx.x;
  if (i >= 1024 * 2048) return;
  int n = i / 2048, k = i % 2048;
  Wnt[i] = (bf16_t)Wn[(size_t)k * 1024 + n];
}

// Two-part padded transpose for Wel/Wnl: W [2*Kpart][Nw] -> Wt [Npad][2*KpadPart]
__global__ void conv_w2t_kernel(const float* __restrict__ W, bf16_t* __restrict__ Wt,
                                int Nw, int Npad, int Kpart, int KpadPart) {
  int total = Npad * 2 * KpadPart;
  int i = blockIdx.x * blockDim.x + threadIdx.x;
  if (i >= total) return;
  int n = i / (2 * KpadPart), k = i % (2 * KpadPart);
  int part = k / KpadPart, kk = k % KpadPart;
  float v = 0.f;
  if (n < Nw && kk < Kpart) v = W[(size_t)(part * Kpart + kk) * Nw + n];
  Wt[i] = (bf16_t)v;
}

__global__ void fill_f32_kernel(float* p, float v, int n) {
  int i = blockIdx.x * blockDim.x + threadIdx.x;
  if (i < n) p[i] = v;
}
__global__ void fill_u32_kernel(unsigned* p, unsigned v, int n) {
  int i = blockIdx.x * blockDim.x + threadIdx.x;
  if (i < n) p[i] = v;
}

// ---------------- edge GEMM: e_f = relu([nf[src]|s_f|nf[dst]] @ We + be) ----------------
// block = 128 thr (4 waves); block tile 64 edges x 64 cols; wave = 16 edges x 64 cols (4 WMMA accs)
// K loop split into 3 branch-free segments so accumulators stay pinned.
__global__ __launch_bounds__(128) void edge_gemm_ef_kernel(
    const bf16_t* __restrict__ nf_bf, const bf16_t* __restrict__ sf_bf,
    const bf16_t* __restrict__ Wet, const float* __restrict__ be,
    const int* __restrict__ src, const int* __restrict__ dst,
    float* __restrict__ e_f) {
  const int lane = threadIdx.x & 31;
  const int wave = threadIdx.x >> 5;
  const int l = lane & 15, hi = lane >> 4;
  const int m0 = blockIdx.y * 64 + wave * 16;
  const int n0 = blockIdx.x * 64;
  const int e = m0 + l;
  const bf16_t* arow_s = nf_bf + (size_t)src[e] * 1024;
  const bf16_t* arow_d = nf_bf + (size_t)dst[e] * 1024;
  const bf16_t* arow_e = sf_bf + (size_t)e * 32;
  const bf16_t* br0 = Wet + (size_t)(n0 +  0 + l) * 2080;
  const bf16_t* br1 = Wet + (size_t)(n0 + 16 + l) * 2080;
  const bf16_t* br2 = Wet + (size_t)(n0 + 32 + l) * 2080;
  const bf16_t* br3 = Wet + (size_t)(n0 + 48 + l) * 2080;
  v8f c0 = {}, c1 = {}, c2 = {}, c3 = {};
#pragma unroll 2
  for (int kb = 0; kb < 1024; kb += 32)           // src-node segment (K' 0..1023)
    gemm_step4(arow_s + kb, br0, br1, br2, br3, kb, hi, c0, c1, c2, c3);
  gemm_step4(arow_e, br0, br1, br2, br3, 1024, hi, c0, c1, c2, c3);  // s_f segment
#pragma unroll 2
  for (int kb = 0; kb < 1024; kb += 32)           // dst-node segment (K' 1056..2079)
    gemm_step4(arow_d + kb, br0, br1, br2, br3, 1056 + kb, hi, c0, c1, c2, c3);

  v8f accs[4] = {c0, c1, c2, c3};
  for (int t = 0; t < 4; ++t) {
    int col = n0 + t * 16 + l;
    float bv = be[col];
    v8f c = accs[t];
#pragma unroll
    for (int i = 0; i < 8; ++i) {
      float v = c[i] + bv; v = v > 0.f ? v : 0.f;
      e_f[(size_t)(m0 + i + hi * 8) * 1024 + col] = v;
    }
  }
}

// ---------------- fused lang edge GEMM + attention score ----------------
// a_feat_lang[e] = relu([w2v[src]|w2v[dst]] @ Wel + bel) . Wal + bal  (e_f_lang never stored)
__global__ __launch_bounds__(128) void edge_gemm_lang_kernel(
    const bf16_t* __restrict__ w2v_bf, const bf16_t* __restrict__ Welt,
    const float* __restrict__ bel, const float* __restrict__ Wal,
    const float* __restrict__ bal,
    const int* __restrict__ src, const int* __restrict__ dst,
    float* __restrict__ afeatl) {
  const int lane = threadIdx.x & 31;
  const int wave = threadIdx.x >> 5;
  const int l = lane & 15, hi = lane >> 4;
  const int m0 = blockIdx.x * 64 + wave * 16;
  const int e = m0 + l;
  const bf16_t* arow_s = w2v_bf + (size_t)src[e] * 320;
  const bf16_t* arow_d = w2v_bf + (size_t)dst[e] * 320;
  float partial[8] = {0, 0, 0, 0, 0, 0, 0, 0};
  for (int nt = 0; nt < 38; ++nt) {               // N = 608 padded (600 real)
    const int col = nt * 16 + l;
    const bf16_t* brow = Welt + (size_t)col * 640;
    v8f c = {};
#pragma unroll 2
    for (int kb = 0; kb < 320; kb += 32)          // w2v[src] segment
      c = WMMA_BF16(load_a_frag(arow_s + kb, hi), load_b_frag(brow + kb, hi), c);
#pragma unroll 2
    for (int kb = 0; kb < 320; kb += 32)          // w2v[dst] segment
      c = WMMA_BF16(load_a_frag(arow_d + kb, hi), load_b_frag(brow + 320 + kb, hi), c);
    float bv  = (col < 600) ? bel[col] : 0.f;
    float wal = (col < 600) ? Wal[col] : 0.f;
#pragma unroll
    for (int i = 0; i < 8; ++i) {
      float v = c[i] + bv; v = v > 0.f ? v : 0.f;
      partial[i] += v * wal;
    }
  }
  float bal0 = bal[0];
#pragma unroll
  for (int i = 0; i < 8; ++i) {
    float s = partial[i];                          // reduce across each 16-lane half
    s += __shfl_xor(s, 1, 32);
    s += __shfl_xor(s, 2, 32);
    s += __shfl_xor(s, 4, 32);
    s += __shfl_xor(s, 8, 32);
    if (l == 0) afeatl[m0 + i + hi * 8] = s + bal0;
  }
}

// ---------------- attention matvec: a_feat = e_f @ Wa + ba ----------------
__global__ __launch_bounds__(256) void afeat_kernel(
    const float* __restrict__ e_f, const float* __restrict__ Wa,
    const float* __restrict__ ba, float* __restrict__ afeat) {
  const int lane = threadIdx.x & 31;
  const int wave = threadIdx.x >> 5;
  const int e = blockIdx.x * 8 + wave;
  const float* row = e_f + (size_t)e * 1024;
  float s = 0.f;
  for (int c = lane; c < 1024; c += 32) s += row[c] * Wa[c];
  s += __shfl_xor(s, 16, 32);
  s += __shfl_xor(s, 8, 32);
  s += __shfl_xor(s, 4, 32);
  s += __shfl_xor(s, 2, 32);
  s += __shfl_xor(s, 1, 32);
  if (lane == 0) afeat[e] = s + ba[0];
}

// ---------------- segment softmax ----------------
__global__ void segmax_kernel(const float* __restrict__ afeat, const float* __restrict__ afeatl,
                              const int* __restrict__ dst,
                              unsigned* __restrict__ menc, unsigned* __restrict__ mencl) {
  int e = blockIdx.x * blockDim.x + threadIdx.x;
  if (e >= N_EDGES) return;
  int d = dst[e];
  atomicMax(&menc[d],  fenc(afeat[e]));
  atomicMax(&mencl[d], fenc(afeatl[e]));
}

__global__ void expdenom_kernel(float* __restrict__ afeat, float* __restrict__ afeatl,
                                const int* __restrict__ dst,
                                const unsigned* __restrict__ menc, const unsigned* __restrict__ mencl,
                                float* __restrict__ denom, float* __restrict__ denoml) {
  int e = blockIdx.x * blockDim.x + threadIdx.x;
  if (e >= N_EDGES) return;
  int d = dst[e];
  float ex = expf(afeat[e] - fdec(menc[d]));
  afeat[e] = ex;
  atomicAdd(&denom[d], ex);
  float exl = expf(afeatl[e] - fdec(mencl[d]));
  afeatl[e] = exl;
  atomicAdd(&denoml[d], exl);
}

// ---------------- weighted scatter: z_f, z_f_lang ----------------
__global__ __launch_bounds__(256) void scatter_kernel(
    const float* __restrict__ nf, const float* __restrict__ w2v,
    const float* __restrict__ e_f,
    const float* __restrict__ ex, const float* __restrict__ exl,
    const float* __restrict__ denom, const float* __restrict__ denoml,
    const int* __restrict__ src, const int* __restrict__ dst,
    float* __restrict__ z_f, float* __restrict__ zfl) {
  const int lane = threadIdx.x & 31;
  const int wave = threadIdx.x >> 5;
  const int e = blockIdx.x * 8 + wave;
  const int s = src[e], d = dst[e];
  const float alpha  = ex[e]  / denom[d];
  const float alphal = exl[e] / denoml[d];
  const float* nr = nf  + (size_t)s * 1024;
  const float* er = e_f + (size_t)e * 1024;
  float* zr = z_f + (size_t)d * 1024;
  for (int c = lane; c < 1024; c += 32)
    atomicAdd(&zr[c], alpha * (nr[c] + er[c]));
  const float* wr = w2v + (size_t)s * 300;
  float* zlr = zfl + (size_t)d * 320;
  for (int c = lane; c < 300; c += 32)
    atomicAdd(&zlr[c], alphal * wr[c]);
}

// ---------------- node GEMM: new_n_f = relu([n_f|z_f] @ Wn + bn) ----------------
__global__ __launch_bounds__(128) void node_gemm_kernel(
    const bf16_t* __restrict__ nf_bf, const bf16_t* __restrict__ zf_bf,
    const bf16_t* __restrict__ Wnt, const float* __restrict__ bn,
    float* __restrict__ out) {
  const int lane = threadIdx.x & 31;
  const int wave = threadIdx.x >> 5;
  const int l = lane & 15, hi = lane >> 4;
  const int m0 = blockIdx.y * 64 + wave * 16;
  const int n0 = blockIdx.x * 64;
  const int row = m0 + l;
  const bf16_t* a0 = nf_bf + (size_t)row * 1024;
  const bf16_t* a1 = zf_bf + (size_t)row * 1024;
  const bf16_t* br0 = Wnt + (size_t)(n0 +  0 + l) * 2048;
  const bf16_t* br1 = Wnt + (size_t)(n0 + 16 + l) * 2048;
  const bf16_t* br2 = Wnt + (size_t)(n0 + 32 + l) * 2048;
  const bf16_t* br3 = Wnt + (size_t)(n0 + 48 + l) * 2048;
  v8f c0 = {}, c1 = {}, c2 = {}, c3 = {};
#pragma unroll 2
  for (int kb = 0; kb < 1024; kb += 32)           // n_f segment
    gemm_step4(a0 + kb, br0, br1, br2, br3, kb, hi, c0, c1, c2, c3);
#pragma unroll 2
  for (int kb = 0; kb < 1024; kb += 32)           // z_f segment
    gemm_step4(a1 + kb, br0, br1, br2, br3, 1024 + kb, hi, c0, c1, c2, c3);

  v8f accs[4] = {c0, c1, c2, c3};
  for (int t = 0; t < 4; ++t) {
    int col = n0 + t * 16 + l;
    float bv = bn[col];
    v8f c = accs[t];
#pragma unroll
    for (int i = 0; i < 8; ++i) {
      float v = c[i] + bv; v = v > 0.f ? v : 0.f;
      out[(size_t)(m0 + i + hi * 8) * 1024 + col] = v;
    }
  }
}

// ---------------- node lang GEMM: new_n_f_lang = relu([w2v|z_f_lang] @ Wnl + bnl) ----------------
__global__ __launch_bounds__(128) void node_gemm_lang_kernel(
    const bf16_t* __restrict__ w2v_bf, const bf16_t* __restrict__ zfl_bf,
    const bf16_t* __restrict__ Wnlt, const float* __restrict__ bnl,
    float* __restrict__ outl) {
  const int lane = threadIdx.x & 31;
  const int wave = threadIdx.x >> 5;
  const int l = lane & 15, hi = lane >> 4;
  const int m0 = blockIdx.x * 64 + wave * 16;
  const int row = m0 + l;
  const bf16_t* a0 = w2v_bf + (size_t)row * 320;
  const bf16_t* a1 = zfl_bf + (size_t)row * 320;
  for (int nt = 0; nt < 19; ++nt) {               // N = 304 padded (300 real)
    const int col = nt * 16 + l;
    const bf16_t* brow = Wnlt + (size_t)col * 640;
    v8f c = {};
#pragma unroll 2
    for (int kb = 0; kb < 320; kb += 32)          // w2v segment
      c = WMMA_BF16(load_a_frag(a0 + kb, hi), load_b_frag(brow + kb, hi), c);
#pragma unroll 2
    for (int kb = 0; kb < 320; kb += 32)          // z_f_lang segment
      c = WMMA_BF16(load_a_frag(a1 + kb, hi), load_b_frag(brow + 320 + kb, hi), c);
    if (col < 300) {
      float bv = bnl[col];
#pragma unroll
      for (int i = 0; i < 8; ++i) {
        float v = c[i] + bv; v = v > 0.f ? v : 0.f;
        outl[(size_t)(m0 + i + hi * 8) * 300 + col] = v;
      }
    }
  }
}

// ---------------- host launcher ----------------
extern "C" void kernel_launch(void* const* d_in, const int* in_sizes, int n_in,
                              void* d_out, int out_size, void* d_ws, size_t ws_size,
                              hipStream_t stream) {
  (void)in_sizes; (void)n_in; (void)out_size; (void)ws_size;
  const float* n_f = (const float*)d_in[0];
  const float* w2v = (const float*)d_in[1];
  const float* s_f = (const float*)d_in[2];
  const int*   src = (const int*)d_in[3];
  const int*   dst = (const int*)d_in[4];
  const float* We  = (const float*)d_in[5];
  const float* be  = (const float*)d_in[6];
  const float* Wel = (const float*)d_in[7];
  const float* bel = (const float*)d_in[8];
  const float* Wa  = (const float*)d_in[9];
  const float* ba  = (const float*)d_in[10];
  const float* Wal = (const float*)d_in[11];
  const float* bal = (const float*)d_in[12];
  const float* Wn  = (const float*)d_in[13];
  const float* bn  = (const float*)d_in[14];
  const float* Wnl = (const float*)d_in[15];
  const float* bnl = (const float*)d_in[16];

  char* ws = (char*)d_ws;
  size_t off = 0;
  auto take = [&](size_t bytes) -> char* {
    char* p = ws + off;
    off = (off + bytes + 255) & ~(size_t)255;
    return p;
  };
  bf16_t*   nf_bf  = (bf16_t*)take((size_t)N_NODES * 1024 * 2);
  bf16_t*   w2v_bf = (bf16_t*)take((size_t)N_NODES * 320 * 2);
  bf16_t*   sf_bf  = (bf16_t*)take((size_t)N_EDGES * 32 * 2);
  bf16_t*   Wet    = (bf16_t*)take((size_t)1024 * 2080 * 2);
  bf16_t*   Welt   = (bf16_t*)take((size_t)608 * 640 * 2);
  bf16_t*   Wnt    = (bf16_t*)take((size_t)1024 * 2048 * 2);
  bf16_t*   Wnlt   = (bf16_t*)take((size_t)304 * 640 * 2);
  float*    e_f    = (float*)take((size_t)N_EDGES * 1024 * 4);
  float*    afeat  = (float*)take((size_t)N_EDGES * 4);
  float*    afeatl = (float*)take((size_t)N_EDGES * 4);
  unsigned* menc   = (unsigned*)take((size_t)N_NODES * 4);
  unsigned* mencl  = (unsigned*)take((size_t)N_NODES * 4);
  float*    denom  = (float*)take((size_t)N_NODES * 4);
  float*    denoml = (float*)take((size_t)N_NODES * 4);
  float*    z_f    = (float*)take((size_t)N_NODES * 1024 * 4);
  float*    zfl    = (float*)take((size_t)N_NODES * 320 * 4);
  bf16_t*   zf_bf  = (bf16_t*)take((size_t)N_NODES * 1024 * 2);
  bf16_t*   zfl_bf = (bf16_t*)take((size_t)N_NODES * 320 * 2);

  float* out_nf  = (float*)d_out;
  float* out_nfl = out_nf + (size_t)N_NODES * 1024;

  const int T = 256;
  auto nb = [](size_t n, int t) { return (unsigned)((n + (size_t)t - 1) / (size_t)t); };

  // 1) precision / layout conversions
  f2bf_kernel<<<nb((size_t)N_NODES * 1024, T), T, 0, stream>>>(n_f, nf_bf, N_NODES * 1024);
  f2bf_pad_kernel<<<nb((size_t)N_NODES * 320, T), T, 0, stream>>>(w2v, w2v_bf, N_NODES, 300, 320);
  f2bf_pad_kernel<<<nb((size_t)N_EDGES * 32, T), T, 0, stream>>>(s_f, sf_bf, N_EDGES, 16, 32);
  conv_wet_kernel<<<nb((size_t)1024 * 2080, T), T, 0, stream>>>(We, Wet);
  conv_w2t_kernel<<<nb((size_t)608 * 640, T), T, 0, stream>>>(Wel, Welt, 600, 608, 300, 320);
  conv_w2t_kernel<<<nb((size_t)304 * 640, T), T, 0, stream>>>(Wnl, Wnlt, 300, 304, 300, 320);
  conv_wnt_kernel<<<nb((size_t)1024 * 2048, T), T, 0, stream>>>(Wn, Wnt);

  // 2) init accumulators (every call: ws is not re-poisoned between replays)
  fill_f32_kernel<<<nb((size_t)N_NODES * 1024, T), T, 0, stream>>>(z_f, 0.f, N_NODES * 1024);
  fill_f32_kernel<<<nb((size_t)N_NODES * 320, T), T, 0, stream>>>(zfl, 0.f, N_NODES * 320);
  fill_f32_kernel<<<nb(N_NODES, T), T, 0, stream>>>(denom, 0.f, N_NODES);
  fill_f32_kernel<<<nb(N_NODES, T), T, 0, stream>>>(denoml, 0.f, N_NODES);
  fill_u32_kernel<<<nb(N_NODES, T), T, 0, stream>>>(menc, 0x007FFFFFu, N_NODES);   // enc(-inf)
  fill_u32_kernel<<<nb(N_NODES, T), T, 0, stream>>>(mencl, 0x007FFFFFu, N_NODES);

  // 3) edge GEMMs (WMMA) + attention scores
  edge_gemm_ef_kernel<<<dim3(16, N_EDGES / 64), 128, 0, stream>>>(
      nf_bf, sf_bf, Wet, be, src, dst, e_f);
  edge_gemm_lang_kernel<<<N_EDGES / 64, 128, 0, stream>>>(
      w2v_bf, Welt, bel, Wal, bal, src, dst, afeatl);
  afeat_kernel<<<N_EDGES / 8, 256, 0, stream>>>(e_f, Wa, ba, afeat);

  // 4) segment softmax
  segmax_kernel<<<nb(N_EDGES, T), T, 0, stream>>>(afeat, afeatl, dst, menc, mencl);
  expdenom_kernel<<<nb(N_EDGES, T), T, 0, stream>>>(afeat, afeatl, dst, menc, mencl, denom, denoml);

  // 5) weighted scatter into z_f / z_f_lang
  scatter_kernel<<<N_EDGES / 8, 256, 0, stream>>>(
      n_f, w2v, e_f, afeat, afeatl, denom, denoml, src, dst, z_f, zfl);

  // 6) convert aggregates to bf16
  f2bf_kernel<<<nb((size_t)N_NODES * 1024, T), T, 0, stream>>>(z_f, zf_bf, N_NODES * 1024);
  f2bf_kernel<<<nb((size_t)N_NODES * 320, T), T, 0, stream>>>(zfl, zfl_bf, N_NODES * 320);

  // 7) node GEMMs (WMMA) -> outputs
  node_gemm_kernel<<<dim3(16, N_NODES / 64), 128, 0, stream>>>(nf_bf, zf_bf, Wnt, bn, out_nf);
  node_gemm_lang_kernel<<<N_NODES / 64, 128, 0, stream>>>(w2v_bf, zfl_bf, Wnlt, bnl, out_nfl);
}
